// ConvLayer_687194768383
// MI455X (gfx1250) — matrix operand order repaired and checked by reading, not compiled, not run on hardware
//
#include <hip/hip_runtime.h>

typedef __attribute__((ext_vector_type(16))) __bf16 v16bf;
typedef __attribute__((ext_vector_type(8)))  __bf16 v8bf;
typedef __attribute__((ext_vector_type(8)))  float  v8f;

#define ELEM   64
#define NBRF   32
#define FAN_IN 160       // 2*ELEM + NBRF
#define COUT   128       // 2*ELEM
#define WT_STRIDE 168    // padded LDS row stride (bf16 elems) for W^T
#define BN_EPS 1e-5f

__device__ __forceinline__ float softplusf(float x) {
    return fmaxf(x, 0.f) + log1pf(expf(-fabsf(x)));
}
__device__ __forceinline__ float sigmoidf(float x) {
    return 1.f / (1.f + expf(-x));
}

// ---------------------------------------------------------------------------
// K0: zero stats + segment-sum accumulator
// ---------------------------------------------------------------------------
__global__ __launch_bounds__(256) void k_zero(float* __restrict__ p, long n) {
    long t = (long)blockIdx.x * 256 + threadIdx.x;
    if (t < n) p[t] = 0.f;
}

// ---------------------------------------------------------------------------
// K1: gather + concat + GEMM (bf16 WMMA, fp32 accum) + bias, store bf16,
//     fused BN1 sum / sumsq accumulation (LDS atomics -> 1 global atomic/blk)
// ---------------------------------------------------------------------------
__global__ __launch_bounds__(256)
void k_gemm_stats(const float* __restrict__ atom,
                  const float* __restrict__ nbrf,
                  const int*   __restrict__ selfIdx,
                  const int*   __restrict__ nbrIdx,
                  const float* __restrict__ W,       // (160,128) row-major
                  const float* __restrict__ bias,    // (128)
                  __bf16*      __restrict__ out1,    // (E,128) bf16
                  float* __restrict__ gsum, float* __restrict__ gsq,
                  int E)
{
    __shared__ __bf16 ldsW[COUT * WT_STRIDE];   // W^T, padded rows
    __shared__ float  ldsSum[COUT];
    __shared__ float  ldsSq[COUT];

    const int t = threadIdx.x;
    // Stage W transposed into B-operand-friendly layout: ldsW[col][k]
    for (int i = t; i < COUT * FAN_IN; i += 256) {
        const int col = i & (COUT - 1);
        const int k   = i >> 7;
        ldsW[col * WT_STRIDE + k] = (__bf16)W[k * COUT + col];
    }
    if (t < COUT) { ldsSum[t] = 0.f; ldsSq[t] = 0.f; }
    __syncthreads();

    const int lane = t & 31;
    const int wave = t >> 5;
    const int hi   = lane >> 4;   // 0: lanes 0-15, 1: lanes 16-31
    const int lr   = lane & 15;

    for (int base = blockIdx.x * 128; base < E; base += gridDim.x * 128) {
        const int e0 = base + wave * 16;        // wave-uniform
        if (e0 >= E) continue;

        // A-matrix row for this lane (lanes L and L+16 read the same row,
        // different K chunks, per the 16-bit A layout)
        const int eA = min(e0 + lr, E - 1);
        const int is = selfIdx[eA];
        const int nn = nbrIdx[eA];
        const float* rowS = atom + (size_t)is * ELEM;
        const float* rowN = atom + (size_t)nn * ELEM;
        const float* rowE = nbrf + (size_t)eA * NBRF;

        v8f acc[8];
        #pragma unroll
        for (int nt = 0; nt < 8; ++nt)
            #pragma unroll
            for (int i = 0; i < 8; ++i) acc[nt][i] = 0.f;

        #pragma unroll
        for (int kk = 0; kk < 5; ++kk) {        // K = 160 = 5 x 32
            const int kb = kk * 32;
            const int k0 = kb + (hi << 3);      // lane's first 8-run
            v16bf a;
            #pragma unroll
            for (int r = 0; r < 2; ++r) {
                const int ks = k0 + r * 16;
                // runs are 8-aligned; concat boundaries (64,128) are too,
                // so a run never straddles segments
                const float* src = (ks < ELEM)     ? rowS + ks
                                 : (ks < 2*ELEM)   ? rowN + (ks - ELEM)
                                                   : rowE + (ks - 2*ELEM);
                const float4 p0 = *(const float4*)(src);
                const float4 p1 = *(const float4*)(src + 4);
                a[r*8+0] = (__bf16)p0.x; a[r*8+1] = (__bf16)p0.y;
                a[r*8+2] = (__bf16)p0.z; a[r*8+3] = (__bf16)p0.w;
                a[r*8+4] = (__bf16)p1.x; a[r*8+5] = (__bf16)p1.y;
                a[r*8+6] = (__bf16)p1.z; a[r*8+7] = (__bf16)p1.w;
            }
            const int kB = kb + (hi << 3);      // B: two 8-runs per lane (table order, matches A)
            #pragma unroll
            for (int nt = 0; nt < 8; ++nt) {
                const int col = nt * 16 + lr;
                const __bf16* wp = ldsW + col * WT_STRIDE + kB;
                union { v16bf v; v8bf h[2]; } bm;
                bm.h[0] = *(const v8bf*)(wp);
                bm.h[1] = *(const v8bf*)(wp + 16);
                acc[nt] = __builtin_amdgcn_wmma_f32_16x16x32_bf16(
                    false, a, false, bm.v, (short)0, acc[nt], false, false);
            }
        }

        // bias add, bf16 store, BN1 stats. C layout: (M = hi*8+i, N = lr)
        #pragma unroll
        for (int nt = 0; nt < 8; ++nt) {
            const int col = nt * 16 + lr;
            const float bb = bias[col];
            float s = 0.f, q = 0.f;
            #pragma unroll
            for (int i = 0; i < 8; ++i) {
                const int e = e0 + hi * 8 + i;
                if (e < E) {
                    const float v = acc[nt][i] + bb;
                    out1[(size_t)e * COUT + col] = (__bf16)v;
                    s += v; q += v * v;
                }
            }
            atomicAdd(&ldsSum[col], s);   // ds_add_f32
            atomicAdd(&ldsSq[col],  q);
        }
    }
    __syncthreads();
    if (t < COUT) {
        unsafeAtomicAdd(&gsum[t], ldsSum[t]);   // global_atomic_add_f32
        unsafeAtomicAdd(&gsq[t],  ldsSq[t]);
    }
}

// ---------------------------------------------------------------------------
// K2/K5: finalize BN stats -> mean, inv_std
// ---------------------------------------------------------------------------
__global__ void k_finalize_bn(const float* __restrict__ gsum,
                              const float* __restrict__ gsq,
                              float* __restrict__ mean,
                              float* __restrict__ istd,
                              float invCount, int nfeat)
{
    const int j = threadIdx.x;
    if (j < nfeat) {
        const float m = gsum[j] * invCount;
        const float v = gsq[j] * invCount - m * m;
        mean[j] = m;
        istd[j] = rsqrtf(v + BN_EPS);
    }
}

// ---------------------------------------------------------------------------
// K3: BN1 apply + sigmoid*softplus gate + segment-sum scatter (fp32 atomics;
//     self_fea_idx is sorted -> high L2 locality on the 12.8MB accumulator)
// ---------------------------------------------------------------------------
__global__ __launch_bounds__(256)
void k_gate_scatter(const __bf16* __restrict__ out1,
                    const int*    __restrict__ selfIdx,
                    const float* __restrict__ g1,  const float* __restrict__ be1,
                    const float* __restrict__ m1,  const float* __restrict__ s1,
                    float* __restrict__ nbrSum, long total /* E*64 */)
{
    long t = (long)blockIdx.x * 256 + threadIdx.x;
    if (t >= total) return;
    const int  j = (int)(t & 63);
    const long e = t >> 6;
    float f = (float)out1[e * COUT + j];
    float c = (float)out1[e * COUT + ELEM + j];
    f = g1[j]        * (f - m1[j])        * s1[j]        + be1[j];
    c = g1[ELEM + j] * (c - m1[ELEM + j]) * s1[ELEM + j] + be1[ELEM + j];
    const float msg = sigmoidf(f) * softplusf(c);
    unsafeAtomicAdd(&nbrSum[(size_t)selfIdx[e] * ELEM + j], msg);
}

// ---------------------------------------------------------------------------
// K4: BN2 stats over (N,64)
// ---------------------------------------------------------------------------
__global__ __launch_bounds__(256)
void k_seg_stats(const float* __restrict__ x,
                 float* __restrict__ gsum, float* __restrict__ gsq,
                 long total /* N*64 */)
{
    __shared__ float sS[ELEM], sQ[ELEM];
    const int t = threadIdx.x;
    if (t < ELEM) { sS[t] = 0.f; sQ[t] = 0.f; }
    __syncthreads();
    float s = 0.f, q = 0.f;
    const long stride = (long)gridDim.x * 256;   // multiple of 64 -> j fixed
    for (long i = (long)blockIdx.x * 256 + t; i < total; i += stride) {
        const float v = x[i];
        s += v; q += v * v;
    }
    const int j = t & 63;
    atomicAdd(&sS[j], s);
    atomicAdd(&sQ[j], q);
    __syncthreads();
    if (t < ELEM) {
        unsafeAtomicAdd(&gsum[t], sS[t]);
        unsafeAtomicAdd(&gsq[t],  sQ[t]);
    }
}

// ---------------------------------------------------------------------------
// K6: BN2 apply + residual + softplus -> output (N,64) fp32
// ---------------------------------------------------------------------------
__global__ __launch_bounds__(256)
void k_output(const float* __restrict__ atom, const float* __restrict__ nbrSum,
              const float* __restrict__ g2, const float* __restrict__ be2,
              const float* __restrict__ m2, const float* __restrict__ s2,
              float* __restrict__ out, long total)
{
    long t = (long)blockIdx.x * 256 + threadIdx.x;
    if (t >= total) return;
    const int j = (int)(t & 63);
    float v = nbrSum[t];
    v = g2[j] * (v - m2[j]) * s2[j] + be2[j];
    out[t] = softplusf(atom[t] + v);
}

// ---------------------------------------------------------------------------
extern "C" void kernel_launch(void* const* d_in, const int* in_sizes, int n_in,
                              void* d_out, int out_size, void* d_ws, size_t ws_size,
                              hipStream_t stream)
{
    const float* atom    = (const float*)d_in[0];
    const float* nbrf    = (const float*)d_in[1];
    const int*   selfIdx = (const int*)d_in[2];
    const int*   nbrIdx  = (const int*)d_in[3];
    const float* W   = (const float*)d_in[4];
    const float* b   = (const float*)d_in[5];
    const float* g1  = (const float*)d_in[6];
    const float* be1 = (const float*)d_in[7];
    const float* g2  = (const float*)d_in[8];
    const float* be2 = (const float*)d_in[9];

    const int N = in_sizes[0] / ELEM;
    const int E = in_sizes[1] / NBRF;

    // workspace layout
    char* base = (char*)d_ws;
    __bf16* out1 = (__bf16*)base;                                   // E*128 bf16
    size_t o = ((size_t)E * COUT * sizeof(__bf16) + 255) & ~(size_t)255;
    float* stats = (float*)(base + o);
    float* gsum1 = stats;         // 128
    float* gsq1  = stats + 128;   // 128
    float* gsum2 = stats + 256;   // 64
    float* gsq2  = stats + 320;   // 64
    float* mean1 = stats + 384;   // 128
    float* istd1 = stats + 512;   // 128
    float* mean2 = stats + 640;   // 64
    float* istd2 = stats + 704;   // 64
    float* nbrSum = stats + 768;  // N*64

    const long ng = (long)N * ELEM;
    const long eg = (long)E * ELEM;
    const long zn = 768 + ng;     // stats + segment accumulator

    hipLaunchKernelGGL(k_zero, dim3((unsigned)((zn + 255) / 256)), dim3(256), 0,
                       stream, stats, zn);

    hipLaunchKernelGGL(k_gemm_stats, dim3(1024), dim3(256), 0, stream,
                       atom, nbrf, selfIdx, nbrIdx, W, b, out1, gsum1, gsq1, E);

    hipLaunchKernelGGL(k_finalize_bn, dim3(1), dim3(128), 0, stream,
                       gsum1, gsq1, mean1, istd1, 1.f / (float)E, COUT);

    hipLaunchKernelGGL(k_gate_scatter, dim3((unsigned)((eg + 255) / 256)), dim3(256),
                       0, stream, out1, selfIdx, g1, be1, mean1, istd1, nbrSum, eg);

    hipLaunchKernelGGL(k_seg_stats, dim3(256), dim3(256), 0, stream,
                       nbrSum, gsum2, gsq2, ng);

    hipLaunchKernelGGL(k_finalize_bn, dim3(1), dim3(64), 0, stream,
                       gsum2, gsq2, mean2, istd2, 1.f / (float)N, ELEM);

    hipLaunchKernelGGL(k_output, dim3((unsigned)((ng + 255) / 256)), dim3(256),
                       0, stream, atom, nbrSum, g2, be2, mean2, istd2,
                       (float*)d_out, ng);
}
